// AetherSparcNet_21268678050178
// MI455X (gfx1250) — matrix operand order repaired and checked by reading, not compiled, not run on hardware
//
#include <hip/hip_runtime.h>
#include <hip/hip_bf16.h>

typedef __attribute__((ext_vector_type(16))) _Float16 v16h;
typedef __attribute__((ext_vector_type(8)))  _Float16 v8h;
typedef __attribute__((ext_vector_type(8)))  float    v8f;

#define HID   32
#define WIN   128
#define ALPHAV 2.5f

// ds_swizzle_b32 XOR butterfly (group-of-32: and_mask=0x1f, or=0, xor=xm)
#define SWZ_XOR(v, xm) \
  __int_as_float(__builtin_amdgcn_ds_swizzle(__float_as_int(v), (((xm) << 10) | 0x1f)))

// Fast math on the serial critical path: v_rcp_f32 instead of IEEE div chains.
__device__ __forceinline__ float fast_rcp(float v) { return __builtin_amdgcn_rcpf(v); }
__device__ __forceinline__ float fast_sigmoid(float x) {
  return fast_rcp(1.0f + __expf(-x));
}
__device__ __forceinline__ float fast_tanh(float x) {
  float e2 = __expf(-2.0f * x);
  return (1.0f - e2) * fast_rcp(1.0f + e2);
}

// ---------------------------------------------------------------------------
// Phase 1: spike mask.  Window is only 128 wide, so each thread sums its own
// 128 x^2 taps from an LDS tile (67 MFLOP total -> noise vs 23.3 TB/s HBM).
// Each wave32 packs its 32 mask bits into one word via ballot (no atomics).
// ---------------------------------------------------------------------------
__global__ void __launch_bounds__(256)
spike_mask_kernel(const float* __restrict__ x, unsigned* __restrict__ maskbits,
                  float* __restrict__ mask_out, int T) {
  __shared__ float tile[256 + WIN];
  const int t0  = blockIdx.x * 256;
  const int tid = threadIdx.x;
  for (int i = tid; i < 256 + WIN; i += 256) {
    int g = t0 - WIN + i;
    tile[i] = (g >= 0 && g < T) ? x[g] : 0.0f;
  }
  __syncthreads();

  const int t = t0 + tid;
  bool m = false;
  if (t < T) {
    int ws  = t - WIN; if (ws < 0) ws = 0;
    int cnt = t - ws;  if (cnt < 1) cnt = 1;
    float s = 0.0f;
#pragma unroll 4
    for (int i = ws; i < t; ++i) {
      float v = tile[i - t0 + WIN];
      s = fmaf(v, v, s);
    }
    float rms = sqrtf(fmaxf(s, 0.0f) / (float)cnt) + 1e-8f;
    float xt  = tile[tid + WIN];
    float xm1 = tile[tid + WIN - 1];
    float xm2 = tile[tid + WIN - 2];
    float err = fabsf(xt - (2.0f * xm1 - xm2));
    m = (err > ALPHAV * rms) || (t < 2);
    mask_out[t] = m ? 1.0f : 0.0f;
  }
  unsigned word = __builtin_amdgcn_ballot_w32(m);
  int wbase = t0 + (tid & ~31);
  if ((tid & 31) == 0 && wbase < T) maskbits[wbase >> 5] = word;
}

// ---------------------------------------------------------------------------
// Phase 2: single-wave event-driven GRU scan.  Latency-bound dependent chain:
// per spike, gh = Whh*h + bhh is 6x v_wmma_f32_16x16x32_f16 (K=32 == HID).
//   A = h replicated across rows (1 ds_store_b16 + 2 ds_load_b128)
//   B = resident f16 tiles of Whh^T (6 x v16h, preloaded once)
//   C = bias-replicated, so bhh rides inside the WMMA.
// D[m][n] = gh[16b+n] for all m -> each lane picks its gate value with one
// cndmask between two accumulator blocks; no post-WMMA shuffling.
// Segment outputs (affine in t) stream out between spikes.
// ---------------------------------------------------------------------------
__global__ void __launch_bounds__(32)
event_gru_kernel(const float* __restrict__ x,
                 const float* __restrict__ Wih, const float* __restrict__ Whh,
                 const float* __restrict__ bih, const float* __restrict__ bhh,
                 const float* __restrict__ Wv,  const float* __restrict__ bv,
                 const float* __restrict__ Ws,  const float* __restrict__ bs,
                 const unsigned* __restrict__ maskbits,
                 float* __restrict__ out, float* __restrict__ outN, int T) {
  const int  lane = threadIdx.x;
  const int  ncol = lane & 15;
  const int  klo  = (lane < 16) ? 0 : 16;
  const bool lo   = (lane < 16);

  // Resident B tiles: B_b[k][n] = Whh[16b+n][k], f16, in contiguous k order (same as A)
  // (lanes 0-15 hold K=0..15, lanes 16-31 hold K=16..31, 2 halves per VGPR).
  v16h B[6];
  v8f  Cb[6];
#pragma unroll
  for (int b = 0; b < 6; ++b) {
    const int row = 16 * b + ncol;
#pragma unroll
    for (int j = 0; j < 16; ++j)
      B[b][j] = (_Float16)Whh[row * HID + klo + j];
    const float bias = bhh[16 * b + ncol];
#pragma unroll
    for (int e = 0; e < 8; ++e) Cb[b][e] = bias;
  }

  // Per-lane input-gate weights (Wih is (3H,2) row-major) and output weights.
  const float wr0 = Wih[(0 * HID + lane) * 2 + 0], wr1 = Wih[(0 * HID + lane) * 2 + 1];
  const float wz0 = Wih[(1 * HID + lane) * 2 + 0], wz1 = Wih[(1 * HID + lane) * 2 + 1];
  const float wn0 = Wih[(2 * HID + lane) * 2 + 0], wn1 = Wih[(2 * HID + lane) * 2 + 1];
  const float br  = bih[lane], bz = bih[HID + lane], bn = bih[2 * HID + lane];
  const float wv  = Wv[lane], wss = Ws[lane];
  const float bvc = bv[0],    bsc = bs[0];

  __shared__ __align__(16) _Float16 hsh[HID];

  float h = 0.0f;
  float a_cur = bvc, b_cur = bsc;
  int   t_prev = 0;
  int   N = 0;

  const int nw = (T + 31) >> 5;
  for (int w = 0; w < nw; ++w) {
    unsigned bits = (unsigned)__builtin_amdgcn_readfirstlane((int)maskbits[w]);
    N += __popc(bits);
    while (bits) {
      const int bit = __ffs(bits) - 1;
      bits &= bits - 1;
      const int ts = (w << 5) + bit;

      // Stream outputs for closed segment [t_prev, ts): affine in t.
      for (int t = t_prev + lane; t < ts; t += 32)
        out[t] = fmaf(b_cur, (float)(t - t_prev) * (1.0f / (float)WIN), a_cur);

      const float xt    = x[ts];
      const float delta = (float)(ts - t_prev) * (1.0f / (float)WIN);

      // Publish h as f16, build A = rows-replicated h in the same contiguous k order as B:
      // lanes 0-15 need K={0..15}, lanes 16-31 need K={16..31}.
      hsh[lane] = (_Float16)h;
      __syncthreads();
      union { v16h v; v8h p[2]; } A;
      A.p[0] = *(const v8h*)&hsh[lo ? 0 : 16];
      A.p[1] = *(const v8h*)&hsh[lo ? 8 : 24];
      __syncthreads();

      v8f d0 = __builtin_amdgcn_wmma_f32_16x16x32_f16(false, A.v, false, B[0], (short)0, Cb[0], false, false);
      v8f d1 = __builtin_amdgcn_wmma_f32_16x16x32_f16(false, A.v, false, B[1], (short)0, Cb[1], false, false);
      v8f d2 = __builtin_amdgcn_wmma_f32_16x16x32_f16(false, A.v, false, B[2], (short)0, Cb[2], false, false);
      v8f d3 = __builtin_amdgcn_wmma_f32_16x16x32_f16(false, A.v, false, B[3], (short)0, Cb[3], false, false);
      v8f d4 = __builtin_amdgcn_wmma_f32_16x16x32_f16(false, A.v, false, B[4], (short)0, Cb[4], false, false);
      v8f d5 = __builtin_amdgcn_wmma_f32_16x16x32_f16(false, A.v, false, B[5], (short)0, Cb[5], false, false);

      // Lane l wants gh[l]: block l>>4 at column l&15 -> one select per gate.
      const float ghr = lo ? d0[0] : d1[0];
      const float ghz = lo ? d2[0] : d3[0];
      const float ghn = lo ? d4[0] : d5[0];

      const float gir = fmaf(wr0, xt, fmaf(wr1, delta, br));
      const float giz = fmaf(wz0, xt, fmaf(wz1, delta, bz));
      const float gin = fmaf(wn0, xt, fmaf(wn1, delta, bn));

      const float r  = fast_sigmoid(gir + ghr);
      const float z  = fast_sigmoid(giz + ghz);
      const float nn = fast_tanh(gin + r * ghn);
      h = fmaf(1.0f - z, nn, z * h);

      // Output projections: full-wave XOR butterflies via ds_swizzle.
      float av = h * wv, bw = h * wss;
      av += SWZ_XOR(av, 16); bw += SWZ_XOR(bw, 16);
      av += SWZ_XOR(av,  8); bw += SWZ_XOR(bw,  8);
      av += SWZ_XOR(av,  4); bw += SWZ_XOR(bw,  4);
      av += SWZ_XOR(av,  2); bw += SWZ_XOR(bw,  2);
      av += SWZ_XOR(av,  1); bw += SWZ_XOR(bw,  1);
      a_cur  = av + bvc;
      b_cur  = bw + bsc;
      t_prev = ts;
    }
  }
  // Tail segment [t_prev, T).
  for (int t = t_prev + lane; t < T; t += 32)
    out[t] = fmaf(b_cur, (float)(t - t_prev) * (1.0f / (float)WIN), a_cur);
  if (lane == 0) outN[0] = (float)N;
}

// ---------------------------------------------------------------------------
extern "C" void kernel_launch(void* const* d_in, const int* in_sizes, int n_in,
                              void* d_out, int out_size, void* d_ws, size_t ws_size,
                              hipStream_t stream) {
  const float* x   = (const float*)d_in[0];
  const float* Wih = (const float*)d_in[1];
  const float* Whh = (const float*)d_in[2];
  const float* bih = (const float*)d_in[3];
  const float* bhh = (const float*)d_in[4];
  const float* Wv  = (const float*)d_in[5];
  const float* bvp = (const float*)d_in[6];
  const float* Ws  = (const float*)d_in[7];
  const float* bsp = (const float*)d_in[8];
  const int T = in_sizes[0];

  float* out   = (float*)d_out;        // T outputs
  float* outN  = out + T;              // scalar N (as float)
  float* maskf = out + T + 1;          // T mask values (0/1)
  unsigned* maskbits = (unsigned*)d_ws;  // (T+31)/32 packed words, fully
                                         // rewritten every launch (no state)

  dim3 g1((unsigned)((T + 255) / 256)), b1(256);
  spike_mask_kernel<<<g1, b1, 0, stream>>>(x, maskbits, maskf, T);
  event_gru_kernel<<<1, 32, 0, stream>>>(x, Wih, Whh, bih, bhh, Wv, bvp, Ws, bsp,
                                         maskbits, out, outN, T);
}
